// Transformer_38792144617900
// MI455X (gfx1250) — matrix operand order repaired and checked by reading, not compiled, not run on hardware
//
#include <hip/hip_runtime.h>
#include <hip/hip_bf16.h>
#include <math.h>

#define B_    4
#define S_    512
#define D_    512
#define H_    8
#define DFF_  2048
#define NLAYER 6
#define NOUT_ 32000
#define DEPTH_ 64

typedef __bf16 bf16_t;
typedef bf16_t v16bf  __attribute__((ext_vector_type(16)));
typedef bf16_t bf16x8 __attribute__((ext_vector_type(8)));
typedef bf16_t bf16x4 __attribute__((ext_vector_type(4)));
typedef float  v8f    __attribute__((ext_vector_type(8)));

// ---------------- async global->LDS copy (CDNA5 ASYNCcnt path), with fallback
#if defined(__has_builtin)
#if __has_builtin(__builtin_amdgcn_global_load_async_to_lds_b128)
#define HAVE_ASYNC_LDS 1
#endif
#endif

#ifdef HAVE_ASYNC_LDS
typedef __attribute__((__vector_size__(4 * sizeof(int)))) int v4i_async;
typedef __attribute__((address_space(1))) v4i_async* gv4ptr;
typedef __attribute__((address_space(3))) v4i_async* lv4ptr;
__device__ __forceinline__ void cp16(const bf16_t* g, bf16_t* l) {
  __builtin_amdgcn_global_load_async_to_lds_b128(
      (gv4ptr)(unsigned long long)g,
      (lv4ptr)(unsigned int)(unsigned long long)l,
      0, 0);
}
__device__ __forceinline__ void wait_async_all() {
#if __has_builtin(__builtin_amdgcn_s_wait_asynccnt)
  __builtin_amdgcn_s_wait_asynccnt(0);
#else
  asm volatile("s_wait_asynccnt 0x0" ::: "memory");
#endif
}
__device__ __forceinline__ void wait_async_4() {   // allow 4 newest copies in flight
#if __has_builtin(__builtin_amdgcn_s_wait_asynccnt)
  __builtin_amdgcn_s_wait_asynccnt(4);
#else
  asm volatile("s_wait_asynccnt 0x4" ::: "memory");
#endif
}
#else
__device__ __forceinline__ void cp16(const bf16_t* g, bf16_t* l) {
  *(uint4*)l = *(const uint4*)g;
}
__device__ __forceinline__ void wait_async_all() {}
__device__ __forceinline__ void wait_async_4() {}
#endif

struct GemmP {
  const bf16_t* A;     // [M,K] row-major bf16
  const bf16_t* Bm;    // [N,K] row-major bf16 (pre-transposed)
  float* C;            // optional fp32 out
  bf16_t* C16;         // optional bf16 out
  const float* bias;   // optional, indexed by n
  int M, N, K;
  int lda, ldb, ldc, ldc16;
  long long sAo, sAi, sBo, sBi, sCo, sCi;   // batch strides (outer/inner), elements
  int inner;
  float alpha;
  int relu;
  int transC16; int tcRows; long long tcBatch;  // transposed bf16 store (per-batch [N,rows])
};

#define BM   128
#define BN   128
#define BKT  32
#define LDSP 40   // bf16 row pitch: 32 + 8 pad -> conflict-free frag reads, 16B aligned

// Double-buffered, async-pipelined 128x128 WMMA GEMM; 8 wave32 waves.
__global__ __launch_bounds__(256) void gemm_bf16_wmma(GemmP p) {
  __shared__ __align__(16) bf16_t sA[2][BM * LDSP];
  __shared__ __align__(16) bf16_t sB[2][BN * LDSP];

  int z  = blockIdx.z;
  int zo = z / p.inner, zi = z % p.inner;
  const bf16_t* A  = p.A  + zo * p.sAo + zi * p.sAi;
  const bf16_t* Bp = p.Bm + zo * p.sBo + zi * p.sBi;
  float*  C   = p.C   ? p.C   + zo * p.sCo + zi * p.sCi : nullptr;
  bf16_t* C16 = p.C16 ? p.C16 + zo * p.sCo + zi * p.sCi : nullptr;

  int m0 = blockIdx.y * BM, n0 = blockIdx.x * BN;
  int tid  = threadIdx.x;
  int lane = tid & 31, wave = tid >> 5;
  int wm   = wave >> 2, wn = wave & 3;     // 2x4 wave grid; each wave 64x32 of C
  int r16  = lane & 15, half = lane >> 4;

  // per-thread copy coordinates: 2 chunks for A, 2 for B (clamped rows keep
  // async issue count deterministic; OOB results are discarded in epilogue)
  int rA[2], cA[2], rB[2], cB[2];
  #pragma unroll
  for (int i = 0; i < 2; i++) {
    int idx = tid + i * 256;
    rA[i] = idx >> 2; cA[i] = (idx & 3) << 3;
    int gm = m0 + rA[i]; if (gm >= p.M) gm = p.M - 1;
    rB[i] = rA[i];       cB[i] = cA[i];
    (void)gm;
  }

  auto issue_tile = [&](int k0, int buf) {
    #pragma unroll
    for (int i = 0; i < 2; i++) {
      int gm = m0 + rA[i]; if (gm >= p.M) gm = p.M - 1;
      cp16(A + (long long)gm * p.lda + k0 + cA[i], &sA[buf][rA[i] * LDSP + cA[i]]);
    }
    #pragma unroll
    for (int i = 0; i < 2; i++) {
      int gn = n0 + rB[i]; if (gn >= p.N) gn = p.N - 1;
      cp16(Bp + (long long)gn * p.ldb + k0 + cB[i], &sB[buf][rB[i] * LDSP + cB[i]]);
    }
  };

  v8f acc[4][2];
  #pragma unroll
  for (int i = 0; i < 4; i++)
    #pragma unroll
    for (int j = 0; j < 2; j++) { v8f zr = {0,0,0,0,0,0,0,0}; acc[i][j] = zr; }

  int nk = p.K / BKT;          // all K here are multiples of 32
  issue_tile(0, 0);
  for (int kt = 0; kt < nk; kt++) {
    int buf = kt & 1;
    if (kt + 1 < nk) {         // prefetch next tile into other buffer, keep it in flight
      issue_tile((kt + 1) * BKT, buf ^ 1);
      wait_async_4();
    } else {
      wait_async_all();
    }
    __syncthreads();

    union FragU { v16bf v; bf16x8 h[2]; };
    FragU fa[4], fb[2];
    #pragma unroll
    for (int im = 0; im < 4; im++) {   // A: lanes0-15 K{0-7,16-23}, lanes16-31 K{8-15,24-31}
      int base = (wm * 64 + im * 16 + r16) * LDSP;
      fa[im].h[0] = *(const bf16x8*)&sA[buf][base + half * 8];
      fa[im].h[1] = *(const bf16x8*)&sA[buf][base + 16 + half * 8];
    }
    #pragma unroll
    for (int in = 0; in < 2; in++) {   // B: lanes0-15 K{0-7,16-23}, lanes16-31 K{8-15,24-31}
      int base = (wn * 32 + in * 16 + r16) * LDSP;
      fb[in].h[0] = *(const bf16x8*)&sB[buf][base + half * 8];
      fb[in].h[1] = *(const bf16x8*)&sB[buf][base + 16 + half * 8];
    }
    #pragma unroll
    for (int im = 0; im < 4; im++)
      #pragma unroll
      for (int in = 0; in < 2; in++)
        acc[im][in] = __builtin_amdgcn_wmma_f32_16x16x32_bf16(
            false, fa[im].v, false, fb[in].v, (short)0, acc[im][in], false, false);
    __syncthreads();
  }

  // epilogue: alpha, bias, relu; fp32 and/or bf16 (optionally transposed) stores
  #pragma unroll
  for (int in = 0; in < 2; in++) {
    int n = n0 + wn * 32 + in * 16 + r16;
    if (n >= p.N) continue;
    float bi = p.bias ? p.bias[n] : 0.f;
    #pragma unroll
    for (int im = 0; im < 4; im++) {
      #pragma unroll
      for (int v = 0; v < 8; v++) {
        int m = m0 + wm * 64 + im * 16 + half * 8 + v;
        if (m >= p.M) continue;
        float r = acc[im][in][v] * p.alpha + bi;
        if (p.relu) r = fmaxf(r, 0.f);
        if (C) C[(long long)m * p.ldc + n] = r;
        if (C16) {
          bf16_t h = (bf16_t)r;
          if (!p.transC16)
            C16[(long long)m * p.ldc16 + n] = h;
          else
            C16[(long long)(m / p.tcRows) * p.tcBatch + (long long)n * p.ldc16 + (m % p.tcRows)] = h;
        }
      }
    }
  }
}

// ---------------- weight convert + transpose: fp32 [K,N] -> bf16 [N,K]
__global__ __launch_bounds__(256) void convert_transpose_w(const float* in, bf16_t* out,
                                                           int K, int N) {
  __shared__ float sh[32][33];
  int k0 = blockIdx.y * 32, n0 = blockIdx.x * 32;
  int tid = threadIdx.x;
  {
    int kk = tid >> 3, nc = (tid & 7) << 2;
    float4 f = make_float4(0.f, 0.f, 0.f, 0.f);
    if (k0 + kk < K && n0 + nc + 4 <= N)
      f = *(const float4*)(in + (long long)(k0 + kk) * N + n0 + nc);
    sh[kk][nc] = f.x; sh[kk][nc + 1] = f.y; sh[kk][nc + 2] = f.z; sh[kk][nc + 3] = f.w;
  }
  __syncthreads();
  {
    int nn = tid >> 3, kc = (tid & 7) << 2;
    if (n0 + nn < N && k0 + kc + 4 <= K) {
      bf16x4 h = { (bf16_t)sh[kc][nn], (bf16_t)sh[kc + 1][nn],
                   (bf16_t)sh[kc + 2][nn], (bf16_t)sh[kc + 3][nn] };
      *(bf16x4*)(out + (long long)(n0 + nn) * K + k0 + kc) = h;
    }
  }
}

// ---------------- softmax with mask; fp32 in, bf16 probs out
__global__ __launch_bounds__(256) void softmax_mask(const float* scores, bf16_t* probs,
                                                    const float* mask, int mode, int S, int H) {
  int q = blockIdx.x;
  int z = blockIdx.y;
  int b = z / H;
  const float* row = scores + (long long)z * S * S + (long long)q * S;
  bf16_t* row16 = probs + (long long)z * S * S + (long long)q * S;
  __shared__ float red[256];
  float vals[8];
  int tid = threadIdx.x;
  int cnt = 0;
  float mx = -1e30f;
  for (int k = tid; k < S; k += 256) {
    float v = row[k];
    if (mode == 1)      v += -1e9f * mask[b * S + k];
    else if (mode == 2) v += -1e9f * mask[q * S + k];
    vals[cnt++] = v;
    mx = fmaxf(mx, v);
  }
  red[tid] = mx; __syncthreads();
  for (int s = 128; s > 0; s >>= 1) { if (tid < s) red[tid] = fmaxf(red[tid], red[tid + s]); __syncthreads(); }
  mx = red[0]; __syncthreads();
  float sum = 0.f; cnt = 0;
  for (int k = tid; k < S; k += 256) { float e = __expf(vals[cnt] - mx); vals[cnt] = e; sum += e; cnt++; }
  red[tid] = sum; __syncthreads();
  for (int s = 128; s > 0; s >>= 1) { if (tid < s) red[tid] += red[tid + s]; __syncthreads(); }
  float inv = 1.f / red[0];
  cnt = 0;
  for (int k = tid; k < S; k += 256) row16[k] = (bf16_t)(vals[cnt++] * inv);
}

// ---------------- y = LayerNorm(x + r)*g + b; dual fp32 + bf16 outputs
__global__ __launch_bounds__(256) void add_layernorm(const float* x, const float* r,
                                                     const float* g, const float* bb,
                                                     float* y, bf16_t* y16, int D) {
  long long base = (long long)blockIdx.x * D;
  int tid = threadIdx.x;
  __shared__ float red[256];
  float vals[8];
  int cnt = 0; float s = 0.f;
  for (int d = tid; d < D; d += 256) { float v = x[base + d] + r[base + d]; vals[cnt++] = v; s += v; }
  red[tid] = s; __syncthreads();
  for (int st = 128; st > 0; st >>= 1) { if (tid < st) red[tid] += red[tid + st]; __syncthreads(); }
  float mean = red[0] / D; __syncthreads();
  s = 0.f; cnt = 0;
  for (int d = tid; d < D; d += 256) { float v = vals[cnt++] - mean; s += v * v; }
  red[tid] = s; __syncthreads();
  for (int st = 128; st > 0; st >>= 1) { if (tid < st) red[tid] += red[tid + st]; __syncthreads(); }
  float inv = rsqrtf(red[0] / D + 1e-6f);
  cnt = 0;
  for (int d = tid; d < D; d += 256) {
    float o = (vals[cnt++] - mean) * inv * g[d] + bb[d];
    y[base + d] = o;
    y16[base + d] = (bf16_t)o;
  }
}

// ---------------- y = x*sqrt(D) + PE[s,d]; dual outputs (PE even-column quirk)
__global__ void embed_pe(const float* x, float* y, bf16_t* y16, int total, int S, int D) {
  int i = blockIdx.x * blockDim.x + threadIdx.x;
  if (i >= total) return;
  int d = i % D;
  int s = (i / D) % S;
  int hd = D >> 1;
  int j = (d < hd) ? d : d - hd;
  float ang = (float)s * __powf(10000.f, -2.f * (float)j / (float)D);
  float pe = (d < hd) ? __sinf(ang) : __cosf(ang);
  float o = x[i] * sqrtf((float)D) + pe;
  y[i] = o;
  y16[i] = (bf16_t)o;
}

// ====================== host-side orchestration ======================
struct DenseW { const float *w, *b; };
struct MhaW   { DenseW q, k, v, o; };
struct LnW    { const float *g, *b; };
struct FfnW   { const float *w1, *b1, *w2, *b2; };
struct MhaB   { bf16_t *q, *k, *v, *o; };
struct FfnB   { bf16_t *w1, *w2; };

static void gemm(hipStream_t st, const bf16_t* A, const bf16_t* Bm,
                 float* C, bf16_t* C16, const float* bias,
                 int M, int N, int K, int lda, int ldb, int ldc, int ldc16,
                 float alpha, int relu,
                 int batches = 1, int inner = 1,
                 long long sAo = 0, long long sAi = 0,
                 long long sBo = 0, long long sBi = 0,
                 long long sCo = 0, long long sCi = 0,
                 int transC16 = 0, int tcRows = 1, long long tcBatch = 0) {
  GemmP p{A, Bm, C, C16, bias, M, N, K, lda, ldb, ldc, ldc16,
          sAo, sAi, sBo, sBi, sCo, sCi, inner, alpha, relu, transC16, tcRows, tcBatch};
  dim3 grid((N + BN - 1) / BN, (M + BM - 1) / BM, batches);
  hipLaunchKernelGGL(gemm_bf16_wmma, grid, dim3(256), 0, st, p);
}

static bf16_t* convw(hipStream_t st, const float* w, int K, int N, char*& cur) {
  bf16_t* out = (bf16_t*)cur;
  cur += (((size_t)K * N * 2 + 255) & ~(size_t)255);
  hipLaunchKernelGGL(convert_transpose_w, dim3((N + 31) / 32, (K + 31) / 32), dim3(256), 0, st,
                     w, out, K, N);
  return out;
}

static void mha(hipStream_t st, const bf16_t* q_in16, const bf16_t* kv_in16,
                const MhaW& w, const MhaB& wb, const float* mask, int mode,
                bf16_t* qb16, bf16_t* kb16, bf16_t* vT16,
                float* sc, bf16_t* sc16, bf16_t* ab16, float* att_out) {
  const int BS = B_ * S_;
  gemm(st, q_in16,  wb.q, nullptr, qb16, w.q.b, BS, D_, D_, D_, D_, 0, D_, 1.f, 0);
  gemm(st, kv_in16, wb.k, nullptr, kb16, w.k.b, BS, D_, D_, D_, D_, 0, D_, 1.f, 0);
  // V projection stored transposed per batch: vT16[b][n][s]
  gemm(st, kv_in16, wb.v, nullptr, vT16, w.v.b, BS, D_, D_, D_, D_, 0, S_, 1.f, 0,
       1, 1, 0, 0, 0, 0, 0, 0, 1, S_, (long long)D_ * S_);
  // scores[b,h] = Q_bh @ K_bh^T / 8 (K already [N,K] per-head view)
  gemm(st, qb16, kb16, sc, nullptr, nullptr, S_, S_, DEPTH_, D_, D_, S_, 0, 0.125f, 0,
       B_ * H_, H_,
       (long long)S_ * D_, (long long)DEPTH_,
       (long long)S_ * D_, (long long)DEPTH_,
       (long long)H_ * S_ * S_, (long long)S_ * S_);
  hipLaunchKernelGGL(softmax_mask, dim3(S_, B_ * H_), dim3(256), 0, st, sc, sc16, mask, mode, S_, H_);
  // O[b,h] = P_bh @ V_bh; V^T is [depth, S] rows -> [N,K] path
  gemm(st, sc16, vT16, nullptr, ab16, nullptr, S_, DEPTH_, S_, S_, S_, 0, D_, 1.f, 0,
       B_ * H_, H_,
       (long long)H_ * S_ * S_, (long long)S_ * S_,
       (long long)D_ * S_, (long long)DEPTH_ * S_,
       (long long)S_ * D_, (long long)DEPTH_);
  gemm(st, ab16, wb.o, att_out, nullptr, w.o.b, BS, D_, D_, D_, D_, D_, 0, 1.f, 0);
}

static void addln(hipStream_t st, const float* x, const float* r, const LnW& w,
                  float* y, bf16_t* y16) {
  hipLaunchKernelGGL(add_layernorm, dim3(B_ * S_), dim3(256), 0, st, x, r, w.g, w.b, y, y16, D_);
}

extern "C" void kernel_launch(void* const* d_in, const int* in_sizes, int n_in,
                              void* d_out, int out_size, void* d_ws, size_t ws_size,
                              hipStream_t stream) {
  (void)in_sizes; (void)n_in; (void)out_size; (void)ws_size;
  int ii = 0;
  auto nf = [&]() { return (const float*)d_in[ii++]; };

  const float* x_in     = nf();
  const float* tar      = nf(); (void)tar;      // unused (reference quirk)
  const float* enc_mask = nf();
  const float* la_mask  = nf();
  const float* dec_mask = nf();

  MhaW emha[NLAYER]; LnW eln1[NLAYER]; FfnW effn[NLAYER]; LnW eln2[NLAYER];
  for (int l = 0; l < NLAYER; l++) {
    emha[l] = MhaW{ DenseW{nf(), nf()}, DenseW{nf(), nf()}, DenseW{nf(), nf()}, DenseW{nf(), nf()} };
    eln1[l] = LnW{nf(), nf()};
    effn[l] = FfnW{nf(), nf(), nf(), nf()};
    eln2[l] = LnW{nf(), nf()};
  }
  MhaW dmha1[NLAYER]; LnW dln1[NLAYER]; MhaW dmha2[NLAYER]; FfnW dffn[NLAYER]; LnW dln3[NLAYER];
  for (int l = 0; l < NLAYER; l++) {
    dmha1[l] = MhaW{ DenseW{nf(), nf()}, DenseW{nf(), nf()}, DenseW{nf(), nf()}, DenseW{nf(), nf()} };
    dln1[l]  = LnW{nf(), nf()};
    dmha2[l] = MhaW{ DenseW{nf(), nf()}, DenseW{nf(), nf()}, DenseW{nf(), nf()}, DenseW{nf(), nf()} };
    dffn[l]  = FfnW{nf(), nf(), nf(), nf()};
    dln3[l]  = LnW{nf(), nf()};
  }
  DenseW outw = DenseW{nf(), nf()};

  // -------- workspace carve
  char* cur = (char*)d_ws;
  auto allocf = [&](size_t n) { float* p = (float*)cur; cur += ((n * 4 + 255) & ~(size_t)255); return p; };
  auto alloch = [&](size_t n) { bf16_t* p = (bf16_t*)cur; cur += ((n * 2 + 255) & ~(size_t)255); return p; };
  const size_t BSD = (size_t)B_ * S_ * D_;

  float*  seq   = allocf(BSD);
  float*  decb  = allocf(BSD);
  float*  att   = allocf(BSD);
  float*  sc    = allocf((size_t)B_ * H_ * S_ * S_);
  bf16_t* seq16 = alloch(BSD);
  bf16_t* dec16 = alloch(BSD);
  bf16_t* qb16  = alloch(BSD);
  bf16_t* kb16  = alloch(BSD);
  bf16_t* vT16  = alloch(BSD);
  bf16_t* ab16  = alloch(BSD);
  bf16_t* ffh16 = alloch((size_t)B_ * S_ * DFF_);
  bf16_t* sc16  = alloch((size_t)B_ * H_ * S_ * S_);

  // -------- one-time weight convert+transpose to bf16 [N,K]
  MhaB emhaB[NLAYER], dmha1B[NLAYER], dmha2B[NLAYER];
  FfnB effnB[NLAYER], dffnB[NLAYER];
  for (int l = 0; l < NLAYER; l++) {
    emhaB[l] = MhaB{ convw(stream, emha[l].q.w, D_, D_, cur), convw(stream, emha[l].k.w, D_, D_, cur),
                     convw(stream, emha[l].v.w, D_, D_, cur), convw(stream, emha[l].o.w, D_, D_, cur) };
    effnB[l] = FfnB{ convw(stream, effn[l].w1, D_, DFF_, cur), convw(stream, effn[l].w2, DFF_, D_, cur) };
  }
  for (int l = 0; l < NLAYER; l++) {
    dmha1B[l] = MhaB{ convw(stream, dmha1[l].q.w, D_, D_, cur), convw(stream, dmha1[l].k.w, D_, D_, cur),
                      convw(stream, dmha1[l].v.w, D_, D_, cur), convw(stream, dmha1[l].o.w, D_, D_, cur) };
    dmha2B[l] = MhaB{ convw(stream, dmha2[l].q.w, D_, D_, cur), convw(stream, dmha2[l].k.w, D_, D_, cur),
                      convw(stream, dmha2[l].v.w, D_, D_, cur), convw(stream, dmha2[l].o.w, D_, D_, cur) };
    dffnB[l] = FfnB{ convw(stream, dffn[l].w1, D_, DFF_, cur), convw(stream, dffn[l].w2, DFF_, D_, cur) };
  }
  bf16_t* outT = convw(stream, outw.w, D_, NOUT_, cur);

  const int BS = B_ * S_;
  const int total = (int)BSD;

  // -------- encoder
  hipLaunchKernelGGL(embed_pe, dim3((total + 255) / 256), dim3(256), 0, stream,
                     x_in, seq, seq16, total, S_, D_);
  for (int l = 0; l < NLAYER; l++) {
    mha(stream, seq16, seq16, emha[l], emhaB[l], enc_mask, 1,
        qb16, kb16, vT16, sc, sc16, ab16, att);
    addln(stream, seq, att, eln1[l], seq, seq16);
    gemm(stream, seq16, effnB[l].w1, nullptr, ffh16, effn[l].b1,
         BS, DFF_, D_, D_, D_, 0, DFF_, 1.f, 1);
    gemm(stream, ffh16, effnB[l].w2, att, nullptr, effn[l].b2,
         BS, D_, DFF_, DFF_, DFF_, D_, 0, 1.f, 0);
    addln(stream, seq, att, eln2[l], seq, seq16);
  }

  // -------- decoder (quirk: pre-processing applied to enc_out; tar unused)
  hipLaunchKernelGGL(embed_pe, dim3((total + 255) / 256), dim3(256), 0, stream,
                     seq, decb, dec16, total, S_, D_);
  for (int l = 0; l < NLAYER; l++) {
    mha(stream, dec16, dec16, dmha1[l], dmha1B[l], la_mask, 2,
        qb16, kb16, vT16, sc, sc16, ab16, att);
    addln(stream, decb, att, dln1[l], decb, dec16);
    mha(stream, dec16, seq16, dmha2[l], dmha2B[l], dec_mask, 1,
        qb16, kb16, vT16, sc, sc16, ab16, att);
    addln(stream, decb, att, dln1[l], decb, dec16);   // quirk: ln1 reused
    gemm(stream, dec16, dffnB[l].w1, nullptr, ffh16, dffn[l].b1,
         BS, DFF_, D_, D_, D_, 0, DFF_, 1.f, 1);
    gemm(stream, ffh16, dffnB[l].w2, att, nullptr, dffn[l].b2,
         BS, D_, DFF_, DFF_, DFF_, D_, 0, 1.f, 0);
    addln(stream, decb, att, dln3[l], decb, dec16);
  }

  // -------- final vocab projection -> d_out [B,S,NOUT] fp32
  gemm(stream, dec16, outT, (float*)d_out, nullptr, outw.b,
       BS, NOUT_, D_, D_, D_, NOUT_, 0, 1.f, 0);
}
